// RWKVTimeMix_21663815041273
// MI455X (gfx1250) — matrix operand order repaired and checked by reading, not compiled, not run on hardware
//
#include <hip/hip_runtime.h>

// ---------------- problem constants (fixed by the reference) ----------------
constexpr int kB = 8;
constexpr int kT = 4096;
constexpr int kC = 1024;
constexpr int kM = kB * kT;          // 32768 rows for every GEMM

// ---------------- vector types ----------------
typedef __attribute__((ext_vector_type(16))) __bf16     v16bf;
typedef __attribute__((ext_vector_type(8)))  float      v8f;
typedef __attribute__((ext_vector_type(4)))  __bf16     bf16x4;
typedef __attribute__((ext_vector_type(4)))  float      f32x4;
typedef __attribute__((ext_vector_type(4)))  unsigned   u32x4;

union FragBF {            // 32 bytes: one WMMA 16-bit A/B fragment per lane
    v16bf v;
    u32x4 q[2];
};

// =====================================================================
// Stage 1: time-shift mix, f32 -> bf16   (xk/xv/xr = mix*x + (1-mix)*xx)
// =====================================================================
__global__ __launch_bounds__(256) void prep_mix_kernel(
    const float* __restrict__ x,
    const float* __restrict__ mk,
    const float* __restrict__ mv,
    const float* __restrict__ mr,
    __bf16* __restrict__ xk,
    __bf16* __restrict__ xv,
    __bf16* __restrict__ xr)
{
    const int C4 = kC / 4;
    size_t i4  = (size_t)blockIdx.x * blockDim.x + threadIdx.x;   // over kM*kC/4
    size_t row = i4 / C4;
    int    c4  = (int)(i4 - row * C4);
    int    t   = (int)(row & (kT - 1));

    const f32x4* x4 = (const f32x4*)x;
    f32x4 xc = x4[i4];
    f32x4 xp = (f32x4)0.0f;
    if (t != 0) xp = x4[i4 - C4];

    f32x4 k4 = ((const f32x4*)mk)[c4];
    f32x4 v4 = ((const f32x4*)mv)[c4];
    f32x4 r4 = ((const f32x4*)mr)[c4];

    bf16x4 ok, ov, orr;
#pragma unroll
    for (int j = 0; j < 4; ++j) {
        ok[j]  = (__bf16)(xc[j] * k4[j] + xp[j] * (1.0f - k4[j]));
        ov[j]  = (__bf16)(xc[j] * v4[j] + xp[j] * (1.0f - v4[j]));
        orr[j] = (__bf16)(xc[j] * r4[j] + xp[j] * (1.0f - r4[j]));
    }
    ((bf16x4*)xk)[i4] = ok;
    ((bf16x4*)xv)[i4] = ov;
    ((bf16x4*)xr)[i4] = orr;
}

// =====================================================================
// Stage 2: weight conversion f32 -> bf16 (Wk|Wv|Wr|Wo concatenated)
// =====================================================================
__global__ __launch_bounds__(256) void convw_kernel(
    const float* __restrict__ Wk, const float* __restrict__ Wv,
    const float* __restrict__ Wr, const float* __restrict__ Wo,
    __bf16* __restrict__ out)
{
    const size_t per = (size_t)kC * kC / 4;                       // f32x4 groups per matrix
    size_t i4 = (size_t)blockIdx.x * blockDim.x + threadIdx.x;    // over 4*per
    int    m  = (int)(i4 / per);
    size_t off = i4 - (size_t)m * per;
    const float* src = (m == 0) ? Wk : (m == 1) ? Wv : (m == 2) ? Wr : Wo;
    f32x4 f = ((const f32x4*)src)[off];
    bf16x4 o;
#pragma unroll
    for (int j = 0; j < 4; ++j) o[j] = (__bf16)f[j];
    ((bf16x4*)out)[i4] = o;
}

// =====================================================================
// Stage 3/5: bf16 GEMM  C[m,n] = sum_k A[m,k] * W[n,k]   (W row-major = B^T)
//   tile 128x128x32, 256 threads = 8 waves (4x2), wave tile 32x64,
//   double-buffered LDS fed by ASYNC global->LDS DMA (ASYNCcnt),
//   WMMA f32_16x16x32_bf16.
//   MODE 0: store f32, MODE 1: sigmoid then store f32.
// =====================================================================
constexpr int BM  = 128;
constexpr int BN  = 128;
constexpr int BK  = 32;
constexpr int LDT = 40;   // LDS row stride (elements): 80B => 16B aligned, 20-bank row rotation

template <int MODE>
__global__ __launch_bounds__(256) void gemm_bf16_kernel(
    const __bf16* __restrict__ A,     // [kM, kC]
    const __bf16* __restrict__ W,     // [kC, kC]
    float* __restrict__ Cout)         // [kM, kC]
{
    __shared__ __bf16 lsA[2][BM * LDT];
    __shared__ __bf16 lsB[2][BN * LDT];

    const int tid   = threadIdx.x;
    const int lane  = tid & 31;
    const int wave  = tid >> 5;
    const int waveM = wave >> 1;                 // 0..3
    const int waveN = wave & 1;                  // 0..1
    const int half  = lane >> 4;                 // 0|1
    const int l16   = lane & 15;

    const int blockRow = blockIdx.y * BM;
    const int blockCol = blockIdx.x * BN;

    const __bf16* Ab = A + (size_t)blockRow * kC;
    const __bf16* Wb = W + (size_t)blockCol * kC;

    // LDS byte addresses for the async DMA (flat addr low 32 bits == LDS offset)
    const unsigned baseA = (unsigned)(size_t)(&lsA[0][0]);
    const unsigned baseB = (unsigned)(size_t)(&lsB[0][0]);
    const unsigned bufBytes = (unsigned)(BM * LDT * 2);   // bytes per buffer stage

    v8f acc[2][4] = {};

    // Issue one K-tile (8 KB A + 8 KB B) as async global->LDS DMA:
    // 512 chunks of 16B per matrix tile, 2 chunks per thread per matrix.
    auto issue = [&](int buf, int k0) {
#pragma unroll
        for (int p = 0; p < 2; ++p) {
            int ch = tid + p * 256;
            int r  = ch >> 2;
            int c8 = (ch & 3) * 8;
            unsigned loff = (unsigned)((r * LDT + c8) * 2) + (unsigned)buf * bufBytes;
            const __bf16* ga = Ab + (size_t)r * kC + k0 + c8;
            const __bf16* gb = Wb + (size_t)r * kC + k0 + c8;
            asm volatile("global_load_async_to_lds_b128 %0, %1, off"
                         :: "v"(baseA + loff), "v"(ga) : "memory");
            asm volatile("global_load_async_to_lds_b128 %0, %1, off"
                         :: "v"(baseB + loff), "v"(gb) : "memory");
        }
    };

    issue(0, 0);

    const int NK = kC / BK;                      // 32 K-steps
    for (int kt = 0; kt < NK; ++kt) {
        // my tile-kt DMA complete (ASYNCcnt in-order for loads)
        asm volatile("s_wait_asynccnt 0" ::: "memory");
        __syncthreads();                          // everyone's tile landed; prev readers done
        if (kt + 1 < NK) {
            issue((kt + 1) & 1, (kt + 1) * BK);   // DMA overlaps compute below
            if (kt + 2 < NK) {                    // warm GL2 for the tile after that
                __builtin_prefetch(Ab + (size_t)(tid >> 1) * kC + (kt + 2) * BK, 0, 1);
                __builtin_prefetch(Wb + (size_t)(tid >> 1) * kC + (kt + 2) * BK, 0, 1);
            }
        }
        const int buf = kt & 1;

        // A fragment: lane<16 -> row m=l16, K {0..7, 16..23}; lane>=16 -> K {8..15, 24..31}
        FragBF fa[2];
#pragma unroll
        for (int mt = 0; mt < 2; ++mt) {
            int row = waveM * 32 + mt * 16 + l16;
            const __bf16* p = &lsA[buf][row * LDT + half * 8];
            fa[mt].q[0] = *(const u32x4*)(p);
            fa[mt].q[1] = *(const u32x4*)(p + 16);
        }
        // B fragment: lane = column n (row of W); lane<16 -> K {0..7, 16..23}; lane>=16 -> K {8..15, 24..31}
        FragBF fb[4];
#pragma unroll
        for (int nt = 0; nt < 4; ++nt) {
            int row = waveN * 64 + nt * 16 + l16;
            const __bf16* p = &lsB[buf][row * LDT + half * 8];
            fb[nt].q[0] = *(const u32x4*)(p);
            fb[nt].q[1] = *(const u32x4*)(p + 16);
        }

#pragma unroll
        for (int mt = 0; mt < 2; ++mt)
#pragma unroll
            for (int nt = 0; nt < 4; ++nt)
                acc[mt][nt] = __builtin_amdgcn_wmma_f32_16x16x32_bf16(
                    false, fa[mt].v, false, fb[nt].v,
                    (short)0, acc[mt][nt], false, false);
    }

    // epilogue: C/D layout — VGPR j: lanes 0-15 row j, lanes 16-31 row j+8; N = lane&15
#pragma unroll
    for (int mt = 0; mt < 2; ++mt)
#pragma unroll
        for (int nt = 0; nt < 4; ++nt)
#pragma unroll
            for (int j = 0; j < 8; ++j) {
                int row = blockRow + waveM * 32 + mt * 16 + j + half * 8;
                int col = blockCol + waveN * 64 + nt * 16 + l16;
                float vv = acc[mt][nt][j];
                if (MODE == 1) vv = 1.0f / (1.0f + __expf(-vv));
                Cout[(size_t)row * kC + col] = vv;
            }
}

// =====================================================================
// Stage 4: sequential WKV scan, one thread per (b,c); writes r*y as bf16
// =====================================================================
__global__ __launch_bounds__(256) void wkv_scan_kernel(
    const float* __restrict__ kk,
    const float* __restrict__ vv,
    const float* __restrict__ rr,
    const float* __restrict__ td,
    const float* __restrict__ tf,
    __bf16* __restrict__ ry)
{
    int tid = blockIdx.x * blockDim.x + threadIdx.x;    // 0 .. kB*kC-1
    int c = tid & (kC - 1);
    int b = tid >> 10;

    const float w = -__expf(td[c]);
    const float u = tf[c];

    float aa = 0.0f, bb = 0.0f, pp = -1e30f;
    size_t idx = (size_t)b * kT * kC + c;
    for (int t = 0; t < kT; ++t, idx += kC) {
        float kt = kk[idx];
        float vt = vv[idx];
        float ww = u + pp;
        float p  = fmaxf(ww, kt);
        float e1 = __expf(ww - p);
        float e2 = __expf(kt - p);
        float y  = (aa * e1 + vt * e2) / (bb * e1 + e2 + 1e-6f);

        float ww2 = w + pp;
        float p2  = fmaxf(ww2, kt);
        float e1b = __expf(ww2 - p2);
        float e2b = __expf(kt - p2);
        aa = aa * e1b + vt * e2b;
        bb = bb * e1b + e2b;
        pp = p2;

        ry[idx] = (__bf16)(rr[idx] * y);
    }
}

// =====================================================================
// launcher
// =====================================================================
extern "C" void kernel_launch(void* const* d_in, const int* in_sizes, int n_in,
                              void* d_out, int out_size, void* d_ws, size_t ws_size,
                              hipStream_t stream)
{
    (void)in_sizes; (void)n_in; (void)out_size; (void)ws_size;

    const float* x  = (const float*)d_in[0];
    const float* td = (const float*)d_in[1];
    const float* tf = (const float*)d_in[2];
    const float* mk = (const float*)d_in[3];
    const float* mv = (const float*)d_in[4];
    const float* mr = (const float*)d_in[5];
    const float* Wk = (const float*)d_in[6];
    const float* Wv = (const float*)d_in[7];
    const float* Wr = (const float*)d_in[8];
    const float* Wo = (const float*)d_in[9];
    float* out = (float*)d_out;

    const size_t MN = (size_t)kM * kC;            // 33.55M
    const size_t CC = (size_t)kC * kC;            // 1.05M

    // workspace layout (f32 k/v/r, bf16 activations, bf16 weights)
    float*  kbuf = (float*)d_ws;
    float*  vbuf = kbuf + MN;
    float*  rbuf = vbuf + MN;
    __bf16* xk   = (__bf16*)(rbuf + MN);
    __bf16* xv   = xk + MN;
    __bf16* xr   = xv + MN;
    __bf16* wbf  = xr + MN;                       // 4*CC bf16
    __bf16* wk_bf = wbf;
    __bf16* wv_bf = wbf + CC;
    __bf16* wr_bf = wbf + 2 * CC;
    __bf16* wo_bf = wbf + 3 * CC;
    __bf16* ry   = xk;                            // reuse: xk consumed before scan

    prep_mix_kernel<<<(unsigned)(MN / 4 / 256), 256, 0, stream>>>(x, mk, mv, mr, xk, xv, xr);
    convw_kernel<<<(unsigned)(4 * CC / 4 / 256), 256, 0, stream>>>(Wk, Wv, Wr, Wo, wbf);

    dim3 g(kC / BN, kM / BM);                     // (8, 256)
    gemm_bf16_kernel<0><<<g, 256, 0, stream>>>(xk, wk_bf, kbuf);
    gemm_bf16_kernel<0><<<g, 256, 0, stream>>>(xv, wv_bf, vbuf);
    gemm_bf16_kernel<1><<<g, 256, 0, stream>>>(xr, wr_bf, rbuf);   // sigmoid fused

    wkv_scan_kernel<<<(kB * kC) / 256, 256, 0, stream>>>(kbuf, vbuf, rbuf, td, tf, ry);

    gemm_bf16_kernel<0><<<g, 256, 0, stream>>>(ry, wo_bf, out);
}
